// NoiseAwareLoss_12902081757647
// MI455X (gfx1250) — matrix operand order repaired and checked by reading, not compiled, not run on hardware
//
#include <hip/hip_runtime.h>
#include <hip/hip_bf16.h>
#include <math.h>

typedef __bf16 v16bf __attribute__((ext_vector_type(16)));
typedef float  v8f   __attribute__((ext_vector_type(8)));

#define NCLS   21
#define CPAD   32
#define DDIM   256
#define HW     (512*512)
#define NTILES (HW/16)            // 16384 tiles of 16 pixels
#define THREADS 256
#define WAVES_PER_BLOCK (THREADS/32)
#define NBLOCKS 256
#define TOTAL_WAVES (NBLOCKS*WAVES_PER_BLOCK)   // 2048 -> exactly 8 tiles/wave
#define WROW   264                // padded bf16 row stride (bank-conflict-free)
#define EPS_N  1e-6f

__global__ __launch_bounds__(THREADS)
void nal_main(const float* __restrict__ y_pred,
              const int*   __restrict__ ycrf,
              const int*   __restrict__ yret,
              const float* __restrict__ feat,
              const float* __restrict__ cw,
              float* __restrict__ partials)
{
    __shared__ __bf16 sWhi[CPAD*WROW];
    __shared__ __bf16 sWlo[CPAD*WROW];
    __shared__ float  sWn[CPAD];
    __shared__ float  sred[THREADS];

    const int tid  = threadIdx.x;
    const int lane = tid & 31;
    const int wave = tid >> 5;
    const int g    = lane >> 4;        // half-wave group (wave32)
    const int col  = lane & 15;        // pixel column / class row index

    // ---- stage classifier weights into LDS as bf16 hi/lo split ----
    for (int idx = tid; idx < CPAD*DDIM; idx += THREADS) {
        int c = idx >> 8;              // /256
        int d = idx & 255;
        float w  = (c < NCLS) ? cw[c*DDIM + d] : 0.0f;
        float hi = __uint_as_float(__float_as_uint(w) & 0xFFFF0000u);
        sWhi[c*WROW + d] = (__bf16)hi;
        sWlo[c*WROW + d] = (__bf16)(w - hi);
    }
    if (tid < CPAD) {
        float s = 0.0f;
        if (tid < NCLS) {
            for (int d = 0; d < DDIM; ++d) { float w = cw[tid*DDIM + d]; s += w*w; }
        }
        sWn[tid] = fmaxf(sqrtf(s), EPS_N);
    }
    __syncthreads();

    float acc_nce = 0.f, acc_dce = 0.f, acc_nw = 0.f, acc_dw = 0.f;

    const int waveGlobal = blockIdx.x * WAVES_PER_BLOCK + wave;

    for (int t = waveGlobal; t < NTILES; t += TOTAL_WAVES) {
        const int p0 = t * 16;
        const int p  = p0 + col;

        v8f acc0 = {};                  // classes 0..15
        v8f acc1 = {};                  // classes 16..31 (>=21 zero-padded)
        float fsq = 0.0f;

        const int c0row = col * WROW;          // A tile0: class = col
        const int c1row = (16 + col) * WROW;   // A tile1: class = 16+col

        // Limited unroll: keeps VGPRs < 256 (no vgpr-msb SALU churn),
        // doubles occupancy while still overlapping two load batches.
        #pragma unroll 2
        for (int k = 0; k < DDIM/32; ++k) {
            const int dB = 32*k + g*16;        // B-operand K base for this lane
            float f[16];
            #pragma unroll
            for (int i = 0; i < 16; ++i)
                f[i] = feat[(size_t)(dB + i) * HW + p];

            v16bf bhi, blo;
            #pragma unroll
            for (int i = 0; i < 16; ++i) {
                float hi = __uint_as_float(__float_as_uint(f[i]) & 0xFFFF0000u);
                bhi[i] = (__bf16)hi;
                blo[i] = (__bf16)(f[i] - hi);
                fsq   += f[i]*f[i];            // exact f32 norm accumulation
            }

            const int dA0 = 32*k + 16*g;       // A elements 0..7
            const int dA1 = dA0 + 8;           // A elements 8..15
            v16bf a0h, a0l, a1h, a1l;
            #pragma unroll
            for (int i = 0; i < 8; ++i) {
                a0h[i]   = sWhi[c0row + dA0 + i];
                a0h[i+8] = sWhi[c0row + dA1 + i];
                a0l[i]   = sWlo[c0row + dA0 + i];
                a0l[i+8] = sWlo[c0row + dA1 + i];
                a1h[i]   = sWhi[c1row + dA0 + i];
                a1h[i+8] = sWhi[c1row + dA1 + i];
                a1l[i]   = sWlo[c1row + dA0 + i];
                a1l[i+8] = sWlo[c1row + dA1 + i];
            }

            acc0 = __builtin_amdgcn_wmma_f32_16x16x32_bf16(false, a0h, false, bhi, (short)0, acc0, false, false);
            acc0 = __builtin_amdgcn_wmma_f32_16x16x32_bf16(false, a0h, false, blo, (short)0, acc0, false, false);
            acc0 = __builtin_amdgcn_wmma_f32_16x16x32_bf16(false, a0l, false, bhi, (short)0, acc0, false, false);
            acc1 = __builtin_amdgcn_wmma_f32_16x16x32_bf16(false, a1h, false, bhi, (short)0, acc1, false, false);
            acc1 = __builtin_amdgcn_wmma_f32_16x16x32_bf16(false, a1h, false, blo, (short)0, acc1, false, false);
            acc1 = __builtin_amdgcn_wmma_f32_16x16x32_bf16(false, a1l, false, bhi, (short)0, acc1, false, false);
        }

        // per-pixel feature norm: lanes l and l^16 cover disjoint halves of D
        float fsqFull = fsq + __shfl_xor(fsq, 16, 32);
        float fnorm   = fmaxf(sqrtf(fsqFull), EPS_N);

        const int yc = ycrf[p];
        const int yr = yret[p];

        // correlation over this lane's classes (exclude padded classes)
        float lmax = 0.0f, lstar = 0.0f;
        #pragma unroll
        for (int r = 0; r < 8; ++r) {
            int c = g*8 + r;                         // tile0: M = 8g + r
            float corr = 1.0f + acc0[r] / (sWn[c] * fnorm);
            lmax = fmaxf(lmax, corr);
            if (c == yc) lstar = corr;
            int c2 = 16 + g*8 + r;                   // tile1
            if (c2 < NCLS) {
                float corr2 = 1.0f + acc1[r] / (sWn[c2] * fnorm);
                lmax = fmaxf(lmax, corr2);
                if (c2 == yc) lstar = corr2;
            }
        }
        float cmax  = fmaxf(lmax, __shfl_xor(lmax, 16, 32));
        float cstar = lstar + __shfl_xor(lstar, 16, 32);
        float ratio = cstar / cmax;
        float conf  = ratio * ratio;                 // GAMMA = 2

        // log-prob sums: this lane covers classes with parity == g
        float slog = 0.0f, crfp = 0.0f;
        for (int c = g; c < NCLS; c += 2) {
            float lp = logf(y_pred[(size_t)c * HW + p]);
            slog += lp;
            if (c == yc) crfp = lp;
        }
        float sumlog = slog + __shfl_xor(slog, 16, 32);
        float logcrf = crfp + __shfl_xor(crfp, 16, 32);

        if (g == 0) {                                // one lane per pixel contributes
            float sf = (yc == yr) ? 1.0f : 0.0f;
            acc_nce += sf * logcrf;
            acc_dce += sf;
            acc_nw  += conf * (sumlog - sf * logcrf);
            acc_dw  += conf * ((float)NCLS - sf);
        }
    }

    // deterministic block tree-reduction of the four partials
    float vals[4] = {acc_nce, acc_dce, acc_nw, acc_dw};
    for (int j = 0; j < 4; ++j) {
        __syncthreads();
        sred[tid] = vals[j];
        __syncthreads();
        for (int s = THREADS/2; s > 0; s >>= 1) {
            if (tid < s) sred[tid] += sred[tid + s];
            __syncthreads();
        }
        if (tid == 0) partials[blockIdx.x*4 + j] = sred[0];
    }
}

__global__ __launch_bounds__(NBLOCKS)
void nal_finalize(const float* __restrict__ partials, float* __restrict__ out)
{
    __shared__ float s0[NBLOCKS], s1[NBLOCKS], s2[NBLOCKS], s3[NBLOCKS];
    const int tid = threadIdx.x;
    s0[tid] = partials[tid*4 + 0];
    s1[tid] = partials[tid*4 + 1];
    s2[tid] = partials[tid*4 + 2];
    s3[tid] = partials[tid*4 + 3];
    __syncthreads();
    for (int s = NBLOCKS/2; s > 0; s >>= 1) {
        if (tid < s) {
            s0[tid] += s0[tid+s]; s1[tid] += s1[tid+s];
            s2[tid] += s2[tid+s]; s3[tid] += s3[tid+s];
        }
        __syncthreads();
    }
    if (tid == 0) {
        float loss_ce  = -s0[0] / s1[0];
        float loss_wce = -s2[0] / s3[0];
        out[0] = loss_ce + 0.1f * loss_wce;
        out[1] = loss_ce;
        out[2] = loss_wce;
    }
}

extern "C" void kernel_launch(void* const* d_in, const int* in_sizes, int n_in,
                              void* d_out, int out_size, void* d_ws, size_t ws_size,
                              hipStream_t stream) {
    const float* y_pred = (const float*)d_in[0];   // [21, 512, 512]
    const int*   ycrf   = (const int*)  d_in[1];   // [512, 512]
    const int*   yret   = (const int*)  d_in[2];   // [512, 512]
    const float* feat   = (const float*)d_in[3];   // [256, 512, 512]
    const float* cw     = (const float*)d_in[4];   // [21, 256]
    float* out      = (float*)d_out;               // 3 scalars
    float* partials = (float*)d_ws;                // NBLOCKS * 4 floats

    nal_main<<<NBLOCKS, THREADS, 0, stream>>>(y_pred, ycrf, yret, feat, cw, partials);
    nal_finalize<<<1, NBLOCKS, 0, stream>>>(partials, out);
}
